// Attention_6674379178757
// MI455X (gfx1250) — matrix operand order repaired and checked by reading, not compiled, not run on hardware
//
#include <hip/hip_runtime.h>

typedef __attribute__((ext_vector_type(16))) __bf16 v16bf;
typedef __attribute__((ext_vector_type(8)))  __bf16 v8bf;
typedef __attribute__((ext_vector_type(8)))  float  v8f;

#define B_   4
#define S_   2048
#define H_   12
#define DM_  768
#define DH_  64
#define IGNORE_ (-100000.0f)

// keep VMEM loads clustered before compute: loads cannot sink past this
#define SCHED_SPLIT() asm volatile("" ::: "memory")

__device__ __forceinline__ v8f zero8() {
  v8f z;
#pragma unroll
  for (int i = 0; i < 8; ++i) z[i] = 0.0f;
  return z;
}

__device__ __forceinline__ v16bf combine16(v8bf lo, v8bf hi) {
  return __builtin_shufflevector(lo, hi, 0, 1, 2, 3, 4, 5, 6, 7,
                                 8, 9, 10, 11, 12, 13, 14, 15);
}

__device__ __forceinline__ v8f wmma_bf16(v16bf a, v16bf b, v8f c) {
  // D = A(16x32 bf16) x B(32x16 bf16) + C(16x16 f32)
  return __builtin_amdgcn_wmma_f32_16x16x32_bf16(false, a, false, b, (short)0, c,
                                                 false, false);
}

// ---- tile loaders (wave32; layouts per CDNA5 ISA §7.12.2) ----
// A tile 16x32 from row-major bf16 (leading dim ld). Per lane: M = lane%16,
// K chunk = 8*(lane/16); elems 0..7 -> K+0..7, elems 8..15 -> K+16..23.
__device__ __forceinline__ v16bf load_a_bf16(const __bf16* p, int ld) {
  const int lane = threadIdx.x & 31;
  const int m  = lane & 15;
  const int kb = (lane >> 4) * 8;
  const __bf16* row = p + (size_t)m * ld + kb;
  v8bf lo = *(const v8bf*)(row);
  v8bf hi = *(const v8bf*)(row + 16);
  return combine16(lo, hi);
}

// B tile 32x16 where memory is N-major (BT[n][kdim], leading dim ld):
// per lane: N = lane%16, K chunk = 8*(lane/16); elems 0..7 -> K+0..7, elems 8..15 -> K+16..23.
__device__ __forceinline__ v16bf load_bT_bf16(const __bf16* p, int ld) {
  const int lane = threadIdx.x & 31;
  const int n  = lane & 15;
  const int kb = (lane >> 4) * 8;
  const __bf16* row = p + (size_t)n * ld + kb;
  v8bf lo = *(const v8bf*)(row);
  v8bf hi = *(const v8bf*)(row + 16);
  return combine16(lo, hi);
}

// ---------------- stage 0: bf16 pre-conversion kernels ----------------
__global__ __launch_bounds__(256) void cvt_x_kernel(const float* __restrict__ x,
                                                    __bf16* __restrict__ xb,
                                                    int n) {
  const int i = blockIdx.x * 256 + threadIdx.x;
  if (i < n) xb[i] = (__bf16)x[i];
}

// W [H][DM][DH] (f32) -> WT [H][DH][DM] (bf16), contiguous writes
__global__ __launch_bounds__(256) void cvt_wT_kernel(const float* __restrict__ W,
                                                     __bf16* __restrict__ WT) {
  const int i = blockIdx.x * 256 + threadIdx.x;
  if (i >= H_ * DH_ * DM_) return;
  const int k = i % DM_;
  const int r = i / DM_;
  const int n = r % DH_;
  const int h = r / DH_;
  WT[i] = (__bf16)W[((size_t)h * DM_ + k) * DH_ + n];
}

// W_O flat [768][768] (f32) -> WOT [768][768] transposed (bf16)
__global__ __launch_bounds__(256) void cvt_woT_kernel(const float* __restrict__ W,
                                                      __bf16* __restrict__ WT) {
  const int i = blockIdx.x * 256 + threadIdx.x;
  if (i >= DM_ * DM_) return;
  const int k = i % DM_;
  const int n = i / DM_;
  WT[i] = (__bf16)W[(size_t)k * DM_ + n];
}

// ---------------- kernel 1: QKV projections ----------------
// one wave computes 16 x 64 q,k,v tiles for one (b,h,s-tile). All 13 load
// pairs of a K-step are fenced ahead of the 12 WMMAs (SCHED_SPLIT) so the
// loads stay clustered in distinct registers -> partial s_wait_loadcnt.
__global__ __launch_bounds__(32, 1) void qkv_proj_kernel(
    const __bf16* __restrict__ xb,
    const __bf16* __restrict__ WqT, const float* __restrict__ bq,
    const __bf16* __restrict__ WkT, const float* __restrict__ bk,
    const __bf16* __restrict__ WvT, const float* __restrict__ bv,
    __bf16* __restrict__ Q, __bf16* __restrict__ K, __bf16* __restrict__ VT) {
  const int st = blockIdx.x & 127;          // 128 s-tiles
  const int h  = (blockIdx.x >> 7) % H_;
  const int b  = blockIdx.x / (128 * H_);
  const int s0 = st * 16;

  const __bf16* xrow = xb + ((size_t)b * S_ + s0) * DM_;
  const __bf16* wq = WqT + (size_t)h * DH_ * DM_;
  const __bf16* wk = WkT + (size_t)h * DH_ * DM_;
  const __bf16* wv = WvT + (size_t)h * DH_ * DM_;

  v8f cq[4], ck[4], cv[4];
#pragma unroll
  for (int t = 0; t < 4; ++t) { cq[t] = zero8(); ck[t] = zero8(); cv[t] = zero8(); }

  for (int kk = 0; kk < DM_; kk += 32) {
    v16bf a = load_a_bf16(xrow + kk, DM_);
    v16bf bq4[4], bk4[4], bv4[4];
#pragma unroll
    for (int t = 0; t < 4; ++t)
      bq4[t] = load_bT_bf16(wq + (size_t)(t * 16) * DM_ + kk, DM_);
#pragma unroll
    for (int t = 0; t < 4; ++t)
      bk4[t] = load_bT_bf16(wk + (size_t)(t * 16) * DM_ + kk, DM_);
#pragma unroll
    for (int t = 0; t < 4; ++t)
      bv4[t] = load_bT_bf16(wv + (size_t)(t * 16) * DM_ + kk, DM_);
    SCHED_SPLIT();  // keep all 26 b128 loads issued ahead of the WMMAs
#pragma unroll
    for (int t = 0; t < 4; ++t) cq[t] = wmma_bf16(a, bq4[t], cq[t]);
#pragma unroll
    for (int t = 0; t < 4; ++t) ck[t] = wmma_bf16(a, bk4[t], ck[t]);
#pragma unroll
    for (int t = 0; t < 4; ++t) cv[t] = wmma_bf16(a, bv4[t], cv[t]);
  }

  const int lane = threadIdx.x & 31;
  const int n16  = lane & 15;
  const int hf   = lane >> 4;
  const size_t bh = (size_t)b * H_ + h;
  __bf16* Qh  = Q  + bh * S_ * DH_;   // [s][64]
  __bf16* Kh  = K  + bh * S_ * DH_;   // [s][64]  (N-major for QK^T B operand)
  __bf16* VTh = VT + bh * DH_ * S_;   // [64][S]  (N-major for P.V B operand)
#pragma unroll
  for (int t = 0; t < 4; ++t) {
    const int n = t * 16 + n16;
    const float bqv = bq[h * DH_ + n];
    const float bkv = bk[h * DH_ + n];
    const float bvv = bv[h * DH_ + n];
#pragma unroll
    for (int j = 0; j < 8; ++j) {
      const int s = s0 + j + 8 * hf;  // C layout: row = j + 8*(lane/16)
      Qh[(size_t)s * DH_ + n]  = (__bf16)(cq[t][j] + bqv);
      Kh[(size_t)s * DH_ + n]  = (__bf16)(ck[t][j] + bkv);
      VTh[(size_t)n * S_ + s]  = (__bf16)(cv[t][j] + bvv);
    }
  }
}

// ---------------- kernel 2: causal flash attention ----------------
// one wave per (b,h,16-row q tile); streams 32-key tiles with online softmax.
// K and V tiles are clustered ahead of compute; softmax VALU work hides the
// V-tile latency.
__global__ __launch_bounds__(32, 1) void attn_kernel(
    const __bf16* __restrict__ Q, const __bf16* __restrict__ K,
    const __bf16* __restrict__ VT, __bf16* __restrict__ Z) {
  __shared__ __bf16 ldsP[16 * 32];  // P tile staging for C->A layout transpose

  const int qt = blockIdx.x & 127;
  const int h  = (blockIdx.x >> 7) % H_;
  const int b  = blockIdx.x / (128 * H_);
  const int q0 = qt * 16;

  const size_t bh = (size_t)b * H_ + h;
  const __bf16* Qh  = Q  + bh * S_ * DH_;
  const __bf16* Kh  = K  + bh * S_ * DH_;
  const __bf16* VTh = VT + bh * DH_ * S_;

  const int lane = threadIdx.x & 31;
  const int n16  = lane & 15;
  const int hf   = lane >> 4;

  // Q tile (16 x 64) held as two A registers: K-dim 0..31 and 32..63
  const v16bf aq0 = load_a_bf16(Qh + (size_t)q0 * DH_, DH_);
  const v16bf aq1 = load_a_bf16(Qh + (size_t)q0 * DH_ + 32, DH_);

  float m[8], l[8];
#pragma unroll
  for (int j = 0; j < 8; ++j) { m[j] = -3.0e38f; l[j] = 0.0f; }
  v8f acc[4];
#pragma unroll
  for (int t = 0; t < 4; ++t) acc[t] = zero8();

  for (int kp0 = 0; kp0 < q0 + 16; kp0 += 32) {  // causal: skip tiles past diag
    // cluster ALL loads of this step: 4 K-tile pairs + 4 V tiles
    v16bf kb0[2], kb1[2], vbt[4];
#pragma unroll
    for (int ht = 0; ht < 2; ++ht) {
      const size_t off = (size_t)(kp0 + ht * 16) * DH_;
      kb0[ht] = load_bT_bf16(Kh + off, DH_);       // d_head dims 0..31
      kb1[ht] = load_bT_bf16(Kh + off + 32, DH_);  // d_head dims 32..63
    }
#pragma unroll
    for (int t = 0; t < 4; ++t)
      vbt[t] = load_bT_bf16(VTh + (size_t)(t * 16) * S_ + kp0, S_);
    SCHED_SPLIT();
    v8f stile[2];
#pragma unroll
    for (int ht = 0; ht < 2; ++ht) {
      v8f sc = zero8();
      sc = wmma_bf16(aq0, kb0[ht], sc);
      sc = wmma_bf16(aq1, kb1[ht], sc);
      const int kcol = kp0 + ht * 16 + n16;
#pragma unroll
      for (int j = 0; j < 8; ++j) {
        const int qrow = q0 + j + 8 * hf;
        const float val = sc[j] * 0.125f;  // 1/sqrt(64)
        sc[j] = (kcol > qrow) ? IGNORE_ : val;
      }
      stile[ht] = sc;
    }
    // online softmax, per C-layout row (reduce across the 16 lanes of a half)
#pragma unroll
    for (int j = 0; j < 8; ++j) {
      float mx = fmaxf(stile[0][j], stile[1][j]);
#pragma unroll
      for (int off = 1; off < 16; off <<= 1)
        mx = fmaxf(mx, __shfl_xor(mx, off, 32));
      const float mnew  = fmaxf(m[j], mx);
      const float alpha = __expf(m[j] - mnew);
      m[j] = mnew;
#pragma unroll
      for (int t = 0; t < 4; ++t) acc[t][j] = acc[t][j] * alpha;
      const float p0 = __expf(stile[0][j] - mnew);
      const float p1 = __expf(stile[1][j] - mnew);
      stile[0][j] = p0;
      stile[1][j] = p1;
      float ls = p0 + p1;
#pragma unroll
      for (int off = 1; off < 16; off <<= 1)
        ls += __shfl_xor(ls, off, 32);
      l[j] = l[j] * alpha + ls;
    }
    // P (16x32, C layout in regs) -> LDS row-major -> reload in A layout
#pragma unroll
    for (int ht = 0; ht < 2; ++ht)
#pragma unroll
      for (int j = 0; j < 8; ++j)
        ldsP[(j + 8 * hf) * 32 + ht * 16 + n16] = (__bf16)stile[ht][j];
    __syncthreads();
    v16bf pa;
    {
      const int mrow = n16;
      const int kb   = hf * 8;
      v8bf lo = *(const v8bf*)&ldsP[mrow * 32 + kb];
      v8bf hi = *(const v8bf*)&ldsP[mrow * 32 + kb + 16];
      pa = combine16(lo, hi);
    }
    __syncthreads();
    // acc += P(16x32) x V(32x64)
#pragma unroll
    for (int t = 0; t < 4; ++t) acc[t] = wmma_bf16(pa, vbt[t], acc[t]);
  }

  __bf16* Zb = Z + ((size_t)b * S_) * DM_;  // Z: [b, s, h*64] for out projection
#pragma unroll
  for (int t = 0; t < 4; ++t) {
    const int n = t * 16 + n16;
#pragma unroll
    for (int j = 0; j < 8; ++j) {
      const int s = q0 + j + 8 * hf;
      Zb[(size_t)s * DM_ + h * DH_ + n] = (__bf16)(acc[t][j] / l[j]);
    }
  }
}

// ---------------- kernel 3: output projection ----------------
// Z[8192,768](bf16) x W_O[768,768] + b_O -> out f32; wave does 16x64 tile
__global__ __launch_bounds__(32, 1) void out_proj_kernel(
    const __bf16* __restrict__ Z, const __bf16* __restrict__ WOT,
    const float* __restrict__ bO, float* __restrict__ out) {
  const int sg = blockIdx.x & 511;  // 512 row tiles of 16
  const int ng = blockIdx.x >> 9;   // 12 column groups of 64
  const int r0 = sg * 16;
  const int n0 = ng * 64;

  v8f acc[4];
#pragma unroll
  for (int t = 0; t < 4; ++t) acc[t] = zero8();

  for (int kk = 0; kk < DM_; kk += 32) {
    v16bf a = load_a_bf16(Z + (size_t)r0 * DM_ + kk, DM_);
    v16bf bt4[4];
#pragma unroll
    for (int t = 0; t < 4; ++t)
      bt4[t] = load_bT_bf16(WOT + (size_t)(n0 + t * 16) * DM_ + kk, DM_);
    SCHED_SPLIT();
#pragma unroll
    for (int t = 0; t < 4; ++t) acc[t] = wmma_bf16(a, bt4[t], acc[t]);
  }

  const int lane = threadIdx.x & 31;
  const int n16  = lane & 15;
  const int hf   = lane >> 4;
#pragma unroll
  for (int t = 0; t < 4; ++t) {
    const int n = n0 + t * 16 + n16;
    const float bias = bO[n];
#pragma unroll
    for (int j = 0; j < 8; ++j) {
      const int r = r0 + j + 8 * hf;
      out[(size_t)r * DM_ + n] = acc[t][j] + bias;
    }
  }
}

extern "C" void kernel_launch(void* const* d_in, const int* in_sizes, int n_in,
                              void* d_out, int out_size, void* d_ws, size_t ws_size,
                              hipStream_t stream) {
  const float* x  = (const float*)d_in[0];
  const float* Wq = (const float*)d_in[1];
  const float* bq = (const float*)d_in[2];
  const float* Wk = (const float*)d_in[3];
  const float* bk = (const float*)d_in[4];
  const float* Wv = (const float*)d_in[5];
  const float* bv = (const float*)d_in[6];
  const float* Wo = (const float*)d_in[7];
  const float* bo = (const float*)d_in[8];
  float* out = (float*)d_out;

  // workspace (bf16): xb, Q, K, VT, Z (6,291,456 each) + WqT/WkT/WvT/WOT
  // (589,824 each) = 67.6 MB total
  const size_t big   = (size_t)B_ * S_ * DM_;   // == B_*H_*S_*DH_
  const size_t small = (size_t)H_ * DH_ * DM_;  // == DM_*DM_
  __bf16* xb  = (__bf16*)d_ws;
  __bf16* Q   = xb + big;
  __bf16* K   = Q + big;
  __bf16* VT  = K + big;
  __bf16* Z   = VT + big;
  __bf16* WqT = Z + big;
  __bf16* WkT = WqT + small;
  __bf16* WvT = WkT + small;
  __bf16* WOT = WvT + small;

  const int nx = (int)big;
  cvt_x_kernel<<<dim3((nx + 255) / 256), dim3(256), 0, stream>>>(x, xb, nx);
  cvt_wT_kernel<<<dim3(((int)small + 255) / 256), dim3(256), 0, stream>>>(Wq, WqT);
  cvt_wT_kernel<<<dim3(((int)small + 255) / 256), dim3(256), 0, stream>>>(Wk, WkT);
  cvt_wT_kernel<<<dim3(((int)small + 255) / 256), dim3(256), 0, stream>>>(Wv, WvT);
  cvt_woT_kernel<<<dim3(((int)small + 255) / 256), dim3(256), 0, stream>>>(Wo, WOT);

  dim3 blk(32);
  qkv_proj_kernel<<<dim3(B_ * H_ * (S_ / 16)), blk, 0, stream>>>(
      xb, WqT, bq, WkT, bk, WvT, bv, Q, K, VT);
  attn_kernel<<<dim3(B_ * H_ * (S_ / 16)), blk, 0, stream>>>(Q, K, VT, Z);
  out_proj_kernel<<<dim3((B_ * S_ / 16) * (DM_ / 64)), blk, 0, stream>>>(
      Z, WOT, bo, out);

  (void)in_sizes; (void)n_in; (void)out_size; (void)ws_size;
}
